// Text2SemanticDecoder_5162550689869
// MI455X (gfx1250) — matrix operand order repaired and checked by reading, not compiled, not run on hardware
//
#include <hip/hip_runtime.h>
#include <hip/hip_bf16.h>
#include <math.h>

#define LAYERS 24
#define BATCH 4
#define SEQ 2048
#define DMODEL 512
#define NHEAD 16
#define DHEAD 32
#define LN_EPS 1e-5f

typedef __attribute__((ext_vector_type(16))) _Float16 v16h;
typedef __attribute__((ext_vector_type(8)))  float    v8f;

__device__ __forceinline__ float wave_sum(float v) {
#pragma unroll
  for (int off = 16; off > 0; off >>= 1) v += __shfl_xor(v, off, 32);
  return v;
}
__device__ __forceinline__ float wave_max(float v) {
#pragma unroll
  for (int off = 16; off > 0; off >>= 1) v = fmaxf(v, __shfl_xor(v, off, 32));
  return v;
}

// ---------------------------------------------------------------------------
// WMMA GEMV: out[b,n] = sum_k xin[b,k] * W[n,k] + bias[n]  (optionally ReLU)
// B=4 rows padded to M=16 WMMA tile. f32 inputs converted to f16 in regs,
// accumulation in f32 via v_wmma_f32_16x16x32_f16.
// Block = 128 threads (4 waves), wave w handles 16 output features.
// ---------------------------------------------------------------------------
__global__ void __launch_bounds__(128)
gemv_wmma_kernel(const float* __restrict__ xin,   // [4, K]
                 const float* __restrict__ W,     // [N, K] row-major
                 const float* __restrict__ bias,  // [N]
                 float* __restrict__ out,         // [4, N]
                 int K, int N, int relu)
{
  const int lane = threadIdx.x & 31;
  const int wv   = threadIdx.x >> 5;
  const int nf0  = (blockIdx.x * 4 + wv) * 16;
  const int m    = lane & 15;   // A row (lanes 0..15 / 16..31 both map M=m)
  const int hi   = lane >> 4;

  // ISA 16-bit A 16x32 layout: lane<16 -> K {0..7, 16..23}; lane>=16 -> {8..15, 24..31}
  const float* xr = xin + (size_t)m * K + hi * 8;
  // B 32x16 layout: lane<16 col n, K {0..7, 16..23}; lane>=16 col n, K {8..15, 24..31}
  const float* wr = W + (size_t)(nf0 + m) * K + hi * 8;

  v8f acc = {};
  for (int k0 = 0; k0 < K; k0 += 32) {
    v16h a = {};
    if (m < BATCH) {
#pragma unroll
      for (int i = 0; i < 8; ++i) a[i]     = (_Float16)xr[k0 + i];
#pragma unroll
      for (int i = 0; i < 8; ++i) a[8 + i] = (_Float16)xr[k0 + 16 + i];
    }
    v16h bm;
#pragma unroll
    for (int i = 0; i < 8; ++i) {
      bm[i]     = (_Float16)__builtin_nontemporal_load(wr + k0 + i);
      bm[8 + i] = (_Float16)__builtin_nontemporal_load(wr + k0 + 16 + i);
    }
    acc = __builtin_amdgcn_wmma_f32_16x16x32_f16(
        /*neg_a=*/false, a, /*neg_b=*/false, bm,
        /*c_mod=*/(short)0, acc, /*reuse_a=*/false, /*reuse_b=*/false);
  }
  // C/D layout: lanes 0..15 hold N=lane, VGPR r = M=r  -> batches 0..3 in acc[0..3]
  if (lane < 16) {
    const int n  = nf0 + lane;
    const float bv = bias[n];
#pragma unroll
    for (int b = 0; b < BATCH; ++b) {
      float v = acc[b] + bv;
      if (relu) v = fmaxf(v, 0.f);
      out[(size_t)b * N + n] = v;
    }
  }
}

// ---------------------------------------------------------------------------
// Scatter new k/v token (from qkv scratch) into output caches at position SEQ.
// ko/vo are layer-base pointers into [B, SEQ+1, D].
// ---------------------------------------------------------------------------
__global__ void scatter_kv_kernel(const float* __restrict__ qkv,
                                  float* __restrict__ ko, float* __restrict__ vo)
{
  int i = blockIdx.x * blockDim.x + threadIdx.x;
  if (i >= BATCH * DMODEL) return;
  int b = i / DMODEL, d = i % DMODEL;
  size_t dst = ((size_t)b * (SEQ + 1) + SEQ) * DMODEL + d;
  ko[dst] = qkv[(size_t)b * 3 * DMODEL + DMODEL + d];
  vo[dst] = qkv[(size_t)b * 3 * DMODEL + 2 * DMODEL + d];
}

// ---------------------------------------------------------------------------
// Fused decode attention + cache copy-out.
// One block per (b,h): 8 waves; lane = head dim, wave strides over positions.
// While streaming k/v it also writes the [0..SEQ) slice of the output caches
// (non-temporal), so the 805MB cache copy costs no extra HBM read pass.
// ---------------------------------------------------------------------------
__global__ void __launch_bounds__(256)
attn_kernel(const float* __restrict__ kc,  // layer base [B, SEQ, D]
            const float* __restrict__ vc,
            const float* __restrict__ qkv, // [B, 3D]
            float* __restrict__ o,         // [B, D]
            float* __restrict__ ko,        // layer base [B, SEQ+1, D]
            float* __restrict__ vo)
{
  const int b    = blockIdx.x >> 4;
  const int h    = blockIdx.x & 15;
  const int tid  = threadIdx.x;
  const int lane = tid & 31;
  const int wv   = tid >> 5;
  const float scale = 0.17677669529663689f;  // 1/sqrt(32)

  __shared__ float sc[SEQ + 1];
  __shared__ float red[8];
  __shared__ float osum[8][32];

  const int    col = h * DHEAD + lane;
  const size_t qb  = (size_t)b * 3 * DMODEL;
  const float  qd  = qkv[qb + col] * scale;
  const size_t kb  = (size_t)b * SEQ * DMODEL + col;        // + p*D
  const size_t ob  = (size_t)b * (SEQ + 1) * DMODEL + col;  // + p*D

  // scores (and copy k slice out)
  for (int p = wv; p <= SEQ; p += 8) {
    float kv;
    if (p < SEQ) {
      kv = __builtin_nontemporal_load(kc + kb + (size_t)p * DMODEL);
      __builtin_nontemporal_store(kv, ko + ob + (size_t)p * DMODEL);
    } else {
      kv = qkv[qb + DMODEL + col];
    }
    float s = wave_sum(qd * kv);
    if (lane == 0) sc[p] = s;
  }
  __syncthreads();

  // softmax max
  float mx = -3.4e38f;
  for (int p = tid; p <= SEQ; p += 256) mx = fmaxf(mx, sc[p]);
  mx = wave_max(mx);
  if (lane == 0) red[wv] = mx;
  __syncthreads();
  mx = red[0];
#pragma unroll
  for (int i = 1; i < 8; ++i) mx = fmaxf(mx, red[i]);
  __syncthreads();  // protect red before reuse

  // exp + sum
  float se = 0.f;
  for (int p = tid; p <= SEQ; p += 256) {
    float e = __expf(sc[p] - mx);
    sc[p] = e;
    se += e;
  }
  se = wave_sum(se);
  if (lane == 0) red[wv] = se;
  __syncthreads();
  float Z = 0.f;
#pragma unroll
  for (int i = 0; i < 8; ++i) Z += red[i];
  const float inv = 1.f / Z;

  // weighted V (and copy v slice out)
  float acc = 0.f;
  for (int p = wv; p <= SEQ; p += 8) {
    float w = sc[p];
    float vvv;
    if (p < SEQ) {
      vvv = __builtin_nontemporal_load(vc + kb + (size_t)p * DMODEL);
      __builtin_nontemporal_store(vvv, vo + ob + (size_t)p * DMODEL);
    } else {
      vvv = qkv[qb + 2 * DMODEL + col];
    }
    acc += w * vvv;
  }
  osum[wv][lane] = acc;
  __syncthreads();
  if (tid < 32) {
    float t = 0.f;
#pragma unroll
    for (int w = 0; w < 8; ++w) t += osum[w][tid];
    o[(size_t)b * DMODEL + h * DHEAD + tid] = t * inv;
  }
}

// ---------------------------------------------------------------------------
// out = LayerNorm(a + r) * w + bias ; one block per batch row, D=512.
// Safe for out == a (each thread re-reads only its own elements).
// ---------------------------------------------------------------------------
__global__ void __launch_bounds__(256)
add_ln_kernel(const float* __restrict__ a, const float* __restrict__ r,
              const float* __restrict__ w, const float* __restrict__ bb,
              float* __restrict__ out)
{
  const int b = blockIdx.x;
  const int tid = threadIdx.x;
  const int lane = tid & 31, wv = tid >> 5;
  __shared__ float red[8];

  const float* ar = a + (size_t)b * DMODEL;
  const float* rr = r + (size_t)b * DMODEL;
  float y0 = ar[tid] + rr[tid];
  float y1 = ar[tid + 256] + rr[tid + 256];

  float s = wave_sum(y0 + y1);
  if (lane == 0) red[wv] = s;
  __syncthreads();
  float tot = 0.f;
#pragma unroll
  for (int i = 0; i < 8; ++i) tot += red[i];
  const float mu = tot * (1.f / DMODEL);
  __syncthreads();

  const float d0 = y0 - mu, d1 = y1 - mu;
  s = wave_sum(d0 * d0 + d1 * d1);
  if (lane == 0) red[wv] = s;
  __syncthreads();
  tot = 0.f;
#pragma unroll
  for (int i = 0; i < 8; ++i) tot += red[i];
  const float rs = rsqrtf(tot * (1.f / DMODEL) + LN_EPS);

  out[(size_t)b * DMODEL + tid]       = d0 * rs * w[tid]       + bb[tid];
  out[(size_t)b * DMODEL + tid + 256] = d1 * rs * w[tid + 256] + bb[tid + 256];
}

// ---------------------------------------------------------------------------
extern "C" void kernel_launch(void* const* d_in, const int* in_sizes, int n_in,
                              void* d_out, int out_size, void* d_ws, size_t ws_size,
                              hipStream_t stream) {
  (void)in_sizes; (void)n_in; (void)out_size; (void)ws_size;
  const float* x       = (const float*)d_in[0];
  const float* k_cache = (const float*)d_in[1];
  const float* v_cache = (const float*)d_in[2];
  const float* qkv_w   = (const float*)d_in[3];
  const float* qkv_b   = (const float*)d_in[4];
  const float* out_w   = (const float*)d_in[5];
  const float* out_b   = (const float*)d_in[6];
  const float* w1      = (const float*)d_in[7];
  const float* b1      = (const float*)d_in[8];
  const float* w2      = (const float*)d_in[9];
  const float* b2      = (const float*)d_in[10];
  const float* ln1w    = (const float*)d_in[11];
  const float* ln1b    = (const float*)d_in[12];
  const float* ln2w    = (const float*)d_in[13];
  const float* ln2b    = (const float*)d_in[14];

  float* out   = (float*)d_out;
  float* out_x = out;                                         // [B,1,D]
  float* out_k = out + (size_t)BATCH * DMODEL;                // [L,B,S+1,D]
  float* out_v = out_k + (size_t)LAYERS * BATCH * (SEQ + 1) * DMODEL;

  float* ws      = (float*)d_ws;
  float* x_cur   = ws;                 // 2048
  float* qkv_s   = x_cur + 2048;       // 6144
  float* attn_o  = qkv_s + 6144;       // 2048
  float* proj_s  = attn_o + 2048;      // 2048
  float* xc_s    = proj_s + 2048;      // 2048
  float* h_s     = xc_s + 2048;        // 8192
  float* proj2_s = h_s + 8192;         // 2048

  hipMemcpyAsync(x_cur, x, (size_t)BATCH * DMODEL * sizeof(float),
                 hipMemcpyDeviceToDevice, stream);

  for (int l = 0; l < LAYERS; ++l) {
    const float* kc = k_cache + (size_t)l * BATCH * SEQ * DMODEL;
    const float* vc = v_cache + (size_t)l * BATCH * SEQ * DMODEL;
    float* ko = out_k + (size_t)l * BATCH * (SEQ + 1) * DMODEL;
    float* vo = out_v + (size_t)l * BATCH * (SEQ + 1) * DMODEL;

    // qkv = x @ qkv_w.T + qkv_b   (N=1536, K=512)
    gemv_wmma_kernel<<<(3 * DMODEL) / 64, 128, 0, stream>>>(
        x_cur, qkv_w + (size_t)l * 3 * DMODEL * DMODEL,
        qkv_b + (size_t)l * 3 * DMODEL, qkv_s, DMODEL, 3 * DMODEL, 0);

    // new token k,v -> output caches at position SEQ
    scatter_kv_kernel<<<(BATCH * DMODEL + 255) / 256, 256, 0, stream>>>(qkv_s, ko, vo);

    // attention (+ streams old cache into output caches)
    attn_kernel<<<BATCH * NHEAD, 256, 0, stream>>>(kc, vc, qkv_s, attn_o, ko, vo);

    // o @ out_w.T + out_b   (N=512, K=512)
    gemv_wmma_kernel<<<DMODEL / 64, 128, 0, stream>>>(
        attn_o, out_w + (size_t)l * DMODEL * DMODEL,
        out_b + (size_t)l * DMODEL, proj_s, DMODEL, DMODEL, 0);

    // xc = LN1(x + proj)
    add_ln_kernel<<<BATCH, 256, 0, stream>>>(
        x_cur, proj_s, ln1w + (size_t)l * DMODEL, ln1b + (size_t)l * DMODEL, xc_s);

    // h = relu(xc @ w1.T + b1)   (N=2048, K=512)
    gemv_wmma_kernel<<<(4 * DMODEL) / 64, 128, 0, stream>>>(
        xc_s, w1 + (size_t)l * 4 * DMODEL * DMODEL,
        b1 + (size_t)l * 4 * DMODEL, h_s, DMODEL, 4 * DMODEL, 1);

    // h @ w2.T + b2   (N=512, K=2048)
    gemv_wmma_kernel<<<DMODEL / 64, 128, 0, stream>>>(
        h_s, w2 + (size_t)l * DMODEL * 4 * DMODEL,
        b2 + (size_t)l * DMODEL, proj2_s, 4 * DMODEL, DMODEL, 0);

    // x = LN2(xc + proj2); final layer writes straight to d_out
    float* xnext = (l == LAYERS - 1) ? out_x : x_cur;
    add_ln_kernel<<<BATCH, 256, 0, stream>>>(
        xc_s, proj2_s, ln2w + (size_t)l * DMODEL, ln2b + (size_t)l * DMODEL, xnext);
  }
}
